// InvariantSelfAttention_87462714016126
// MI455X (gfx1250) — compile-verified, matrix operand order checked by reading, not run on hardware
//
#include <hip/hip_runtime.h>
#include <hip/hip_bf16.h>
#include <stdint.h>

#define BB 4
#define TT 1024
#define EE 512
#define SS 1536
#define HH 1024
#define NHEAD 16
#define HD 64
#define NEGINF -1.0e30f

typedef __bf16 bf16_t;
typedef __attribute__((ext_vector_type(16))) __bf16 bf16x16;
typedef __attribute__((ext_vector_type(8)))  float  floatx8;
typedef __attribute__((ext_vector_type(4)))  uint32_t u32x4;
typedef __attribute__((ext_vector_type(8)))  int32_t  i32x8;
typedef __attribute__((ext_vector_type(4)))  int32_t  i32x4;

__device__ __forceinline__ bf16_t f2bf(float x) {
  union { float f; uint32_t u; } v; v.f = x;
  uint32_t r = v.u + 0x7fffu + ((v.u >> 16) & 1u);   // round-to-nearest-even
  uint16_t h = (uint16_t)(r >> 16);
  return __builtin_bit_cast(bf16_t, h);
}

__device__ __forceinline__ floatx8 wmma_bf16(bf16x16 a, bf16x16 b, floatx8 c) {
  return __builtin_amdgcn_wmma_f32_16x16x32_bf16(false, a, false, b, (short)0, c,
                                                 false, false);
}

// A/B fragment (16x32 bf16, wave32 layout) from bf16 row-major source.
// lane L: row/col = base + (L&15); K pattern: half*8+0..7 then 16+half*8+0..7
__device__ __forceinline__ bf16x16 frag_from_bf16(const bf16_t* base, int ld, int row,
                                                  int k0, int half) {
  bf16x16 f;
  const bf16_t* p = base + (size_t)row * ld + k0 + half * 8;
#pragma unroll
  for (int j = 0; j < 8; ++j) f[j] = p[j];
#pragma unroll
  for (int j = 0; j < 8; ++j) f[8 + j] = p[16 + j];
  return f;
}

// B-fragment of a 32x64 chunk (keys x headdim): lane column n, strided K reads.
__device__ __forceinline__ bf16x16 load_vfrag(const bf16_t* vbase, int n, int half) {
  bf16x16 f;
  const bf16_t* p = vbase + (size_t)(half * 8) * HD + n;
#pragma unroll
  for (int j = 0; j < 8; ++j) f[j] = p[(size_t)j * HD];
  const bf16_t* p2 = p + (size_t)16 * HD;
#pragma unroll
  for (int j = 0; j < 8; ++j) f[8 + j] = p2[(size_t)j * HD];
  return f;
}

__device__ __forceinline__ uint32_t lds_off(const void* p) {
  return (uint32_t)(uintptr_t)p;   // low 32 bits of the LDS aperture address
}

// ---------------------------------------------------------------------------
// TDM: async DMA of a [32 rows x 64 bf16] tile (row stride 64 elems) -> LDS.
// D# group0/group1 built per CDNA5 ISA.
// ---------------------------------------------------------------------------
__device__ __forceinline__ void tdm_load_tile32x64(uint32_t ldsaddr, const bf16_t* src) {
  const uint64_t ga = (uint64_t)(uintptr_t)src;
  u32x4 g0;
  g0[0] = 1u;                                               // count=1, user D#
  g0[1] = ldsaddr;                                          // lds_addr (bytes)
  g0[2] = (uint32_t)ga;                                     // global_addr[31:0]
  g0[3] = (uint32_t)((ga >> 32) & 0x01FFFFFFu) | (2u << 30);// addr[56:32] | type=2
  i32x8 g1;
  g1[0] = (int32_t)(1u << 16);        // wg_mask=0, data_size=1 (2 bytes)
  g1[1] = (int32_t)(64u << 16);       // tensor_dim0[15:0]=64 (bits 63:48)
  g1[2] = (int32_t)(32u << 16);       // dim0 hi=0 | tensor_dim1[15:0]=32
  g1[3] = (int32_t)(64u << 16);       // dim1 hi=0 | tile_dim0=64
  g1[4] = 32;                         // tile_dim1=32 | tile_dim2=0
  g1[5] = 64;                         // tensor_dim0_stride[31:0]=64
  g1[6] = 0;                          // stride0 hi | stride1 lo
  g1[7] = 0;                          // stride1 hi
  const i32x4 z4 = {};
#if __clang_major__ >= 23
  const i32x8 z8 = {};
  __builtin_amdgcn_tensor_load_to_lds(g0, g1, z4, z4, z8, 0);
#else
  __builtin_amdgcn_tensor_load_to_lds(g0, g1, z4, z4, 0);
#endif
}

// ---------------------------------------------------------------------------
// Kernel 0: one-time fp32 -> bf16 conversion (x and the four weight matrices)
// ---------------------------------------------------------------------------
__global__ void f32_to_bf16_kernel(const float* __restrict__ in, bf16_t* __restrict__ out,
                                   int n) {
  const int i = (blockIdx.x * 256 + threadIdx.x) * 4;
  if (i + 3 < n) {
    const float4 v = *(const float4*)(in + i);
    out[i + 0] = f2bf(v.x); out[i + 1] = f2bf(v.y);
    out[i + 2] = f2bf(v.z); out[i + 3] = f2bf(v.w);
  }
}

// ---------------------------------------------------------------------------
// Kernel 1: fused QKV projection, bf16 operands, fp32 accumulate.
// grid (H/256, M/16, 3), block (32,4): each wave -> 16x64 strip (4 tiles),
// A fragment loaded once per k-step and reused across 4 WMMAs.
// ---------------------------------------------------------------------------
__global__ void qkv_gemm_kernel(const bf16_t* __restrict__ x,
                                const bf16_t* __restrict__ wq,
                                const bf16_t* __restrict__ wk,
                                const bf16_t* __restrict__ wv,
                                float* __restrict__ qf, float* __restrict__ kf,
                                float* __restrict__ vf) {
  const int lane = threadIdx.x, m16 = lane & 15, half = lane >> 4;
  const int tileN0 = (blockIdx.x * 4 + threadIdx.y) * 64;
  const int tileM = blockIdx.y * 16;
  const int z = blockIdx.z;
  const bf16_t* w = (z == 0) ? wq : (z == 1) ? wk : wv;
  float* out = (z == 0) ? qf : (z == 1) ? kf : vf;
  const float scale = (z == 0) ? 0.125f : 1.0f;  // SCALING = sqrt(64)/64
  floatx8 acc[4] = {};
  for (int k0 = 0; k0 < HH; k0 += 32) {
    const bf16x16 a = frag_from_bf16(x, HH, tileM + m16, k0, half);
#pragma unroll
    for (int j = 0; j < 4; ++j) {
      bf16x16 b = frag_from_bf16(w, HH, tileN0 + j * 16 + m16, k0, half);
      acc[j] = wmma_bf16(a, b, acc[j]);
    }
  }
#pragma unroll
  for (int j = 0; j < 4; ++j)
#pragma unroll
    for (int r = 0; r < 8; ++r)
      out[(size_t)(tileM + r + 8 * half) * HH + tileN0 + j * 16 + m16] = acc[j][r] * scale;
}

// ---------------------------------------------------------------------------
// Kernel 2: RoPE(q) + head pack -> bf16 [B,NH,T,HD]
// ---------------------------------------------------------------------------
__global__ void rope_q_kernel(const float* __restrict__ qf, bf16_t* __restrict__ qh) {
  const int idx = blockIdx.x * 256 + threadIdx.x;   // exact grid
  const int hh = idx & (HH - 1);
  const int t  = (idx >> 10) & (TT - 1);
  const int b  = idx >> 20;
  const int head = hh >> 6, d = hh & 63;
  const float xv = qf[idx];
  const int i = d & 31;
  const float inv = __powf(10000.0f, -(float)(2 * i) * (1.0f / 64.0f));
  const float ang = (float)t * inv;
  const float c = __cosf(ang), s = __sinf(ang);
  const int pd = (d < 32) ? d + 32 : d - 32;
  const float xp = qf[(idx & ~63) + pd];
  const float v = (d < 32) ? (xv * c - xp * s) : (xv * c + xp * s);
  qh[(((size_t)b * NHEAD + head) * TT + t) * HD + d] = f2bf(v);
}

// ---------------------------------------------------------------------------
// Kernel 3: PBC gather + RoPE(k at expanded position) + head pack -> [B,NH,S,HD]
// ---------------------------------------------------------------------------
__global__ void rope_kv_kernel(const float* __restrict__ kf, const float* __restrict__ vf,
                               const int* __restrict__ outcell,
                               bf16_t* __restrict__ kh, bf16_t* __restrict__ vh) {
  const int idx = blockIdx.x * 256 + threadIdx.x;   // exact grid
  const int hh = idx & (HH - 1);
  const int rest = idx >> 10;                        // b*S + s
  const int s = rest % SS;
  const int b = rest / SS;
  const int head = hh >> 6, d = hh & 63;
  const int src = (s < TT) ? s : outcell[b * EE + (s - TT)];
  const size_t rowoff = ((size_t)b * TT + src) * HH + (hh & ~63);
  const float kx = kf[rowoff + d];
  const float vx = vf[rowoff + d];
  const int i = d & 31;
  const float inv = __powf(10000.0f, -(float)(2 * i) * (1.0f / 64.0f));
  const float ang = (float)s * inv;                  // RoPE uses expanded position
  const float c = __cosf(ang), sn = __sinf(ang);
  const int pd = (d < 32) ? d + 32 : d - 32;
  const float kp = kf[rowoff + pd];
  const float kv = (d < 32) ? (kx * c - kp * sn) : (kx * c + kp * sn);
  const size_t o = (((size_t)b * NHEAD + head) * SS + s) * HD + d;
  kh[o] = f2bf(kv);
  vh[o] = f2bf(vx);
}

// ---------------------------------------------------------------------------
// Kernel 4: fused flash attention with TDM double-buffered K/V staging.
// grid (T/64, NH, B), block (32,4). Wave threadIdx.y==0 issues the DMA; all
// 4 q-tile waves consume the shared LDS copy (4x less L2 traffic than per-wave
// loads). probs = softmax(QK^T + bias, masked) * local_w (numerator only).
// ---------------------------------------------------------------------------
__global__ void attention_kernel(const bf16_t* __restrict__ qh, const bf16_t* __restrict__ kh,
                                 const bf16_t* __restrict__ vh,
                                 const float* __restrict__ bias, const float* __restrict__ lw,
                                 const uint8_t* __restrict__ kpm, const uint8_t* __restrict__ em,
                                 float* __restrict__ attn_out) {
  const int lane = threadIdx.x, m16 = lane & 15, half = lane >> 4;
  const int wvid = threadIdx.y;
  const int qtile = (blockIdx.x * 4 + wvid) * 16;
  const int h = blockIdx.y, b = blockIdx.z;

  __shared__ bf16_t kbuf[2][32 * HD];     // double-buffered K chunk (4 KB each)
  __shared__ bf16_t vbuf[2][32 * HD];     // double-buffered V chunk
  __shared__ bf16_t pbuf[4][16 * 32];     // per-wave P staging
  bf16_t* lp = &pbuf[wvid][0];

  const bf16_t* qbase = qh + ((size_t)(b * NHEAD + h) * TT) * HD;
  const bf16_t* kbase = kh + ((size_t)(b * NHEAD + h) * SS) * HD;
  const bf16_t* vbase = vh + ((size_t)(b * NHEAD + h) * SS) * HD;
  const float* bbase = bias + ((size_t)(b * NHEAD + h) * TT) * SS;
  const float* wbase = lw + ((size_t)b * TT) * SS;

  const bf16x16 qa0 = frag_from_bf16(qbase, HD, qtile + m16, 0, half);
  const bf16x16 qa1 = frag_from_bf16(qbase, HD, qtile + m16, 32, half);

  floatx8 o0 = {}, o1 = {}, o2 = {}, o3 = {};
  float mrow[8], lrow[8];
#pragma unroll
  for (int r = 0; r < 8; ++r) { mrow[r] = NEGINF; lrow[r] = 0.0f; }

  // warm the pipeline: chunks 0 and 1 in flight (4 TDM ops)
  if (wvid == 0) {
    tdm_load_tile32x64(lds_off(&kbuf[0][0]), kbase);
    tdm_load_tile32x64(lds_off(&vbuf[0][0]), vbase);
    tdm_load_tile32x64(lds_off(&kbuf[1][0]), kbase + 32 * HD);
    tdm_load_tile32x64(lds_off(&vbuf[1][0]), vbase + 32 * HD);
  }

  const int NCH = SS / 32;                // 48 chunks
  for (int it = 0; it < NCH; ++it) {
    if (wvid == 0) {
      if (it + 1 < NCH) __builtin_amdgcn_s_wait_tensorcnt(2);  // chunk it landed
      else              __builtin_amdgcn_s_wait_tensorcnt(0);
    }
    __syncthreads();                      // hand LDS chunk to all waves
    const bf16_t* kc = &kbuf[it & 1][0];
    const bf16_t* vc = &vbuf[it & 1][0];
    const int s0 = it * 32;

    // ---- scores 16x32: two 16x16 WMMA pairs (K split over headdim) ----
    floatx8 sacc[2];
#pragma unroll
    for (int ks = 0; ks < 2; ++ks) {
      bf16x16 kb0 = frag_from_bf16(kc, HD, ks * 16 + m16, 0, half);
      bf16x16 kb1 = frag_from_bf16(kc, HD, ks * 16 + m16, 32, half);
      floatx8 t = {};
      t = wmma_bf16(qa0, kb0, t);
      t = wmma_bf16(qa1, kb1, t);
      sacc[ks] = t;
    }
    // ---- bias + masks + local weight ----
    float sc[2][8], wl[2][8];
#pragma unroll
    for (int ks = 0; ks < 2; ++ks) {
      const int skey = s0 + ks * 16 + m16;
      const bool km = (skey < TT) ? (kpm[b * TT + skey] != 0)
                                  : (em[b * EE + (skey - TT)] != 0);
#pragma unroll
      for (int r = 0; r < 8; ++r) {
        const int q = qtile + r + 8 * half;
        const float w = wbase[(size_t)q * SS + skey];
        const float s = sacc[ks][r] + bbase[(size_t)q * SS + skey];
        const bool masked = km || (w <= 1e-5f);
        sc[ks][r] = masked ? NEGINF : s;
        wl[ks][r] = w;
      }
    }
    // ---- online softmax: row reductions across the 16-lane half-group ----
    float mloc[8];
#pragma unroll
    for (int r = 0; r < 8; ++r) mloc[r] = fmaxf(sc[0][r], sc[1][r]);
#pragma unroll
    for (int off = 1; off <= 8; off <<= 1)
#pragma unroll
      for (int r = 0; r < 8; ++r) mloc[r] = fmaxf(mloc[r], __shfl_xor(mloc[r], off, 32));

    float alpha[8], psum[8], p[2][8];
#pragma unroll
    for (int r = 0; r < 8; ++r) {
      const float mnew = fmaxf(mrow[r], mloc[r]);
      alpha[r] = __expf(mrow[r] - mnew);
      mrow[r] = mnew;
      const float p0 = (sc[0][r] <= 0.5f * NEGINF) ? 0.0f : __expf(sc[0][r] - mnew);
      const float p1 = (sc[1][r] <= 0.5f * NEGINF) ? 0.0f : __expf(sc[1][r] - mnew);
      p[0][r] = p0; p[1][r] = p1;
      psum[r] = p0 + p1;
    }
#pragma unroll
    for (int off = 1; off <= 8; off <<= 1)
#pragma unroll
      for (int r = 0; r < 8; ++r) psum[r] += __shfl_xor(psum[r], off, 32);
#pragma unroll
    for (int r = 0; r < 8; ++r) lrow[r] = lrow[r] * alpha[r] + psum[r];

    // ---- stage weighted P (C-layout -> row-major LDS) for A-fragment reload ----
#pragma unroll
    for (int ks = 0; ks < 2; ++ks)
#pragma unroll
      for (int r = 0; r < 8; ++r)
        lp[(r + 8 * half) * 32 + ks * 16 + m16] = f2bf(p[ks][r] * wl[ks][r]);
    asm volatile("s_wait_dscnt 0" ::: "memory");   // CDNA5 split DS counter

#pragma unroll
    for (int r = 0; r < 8; ++r) {
      o0[r] *= alpha[r]; o1[r] *= alpha[r]; o2[r] *= alpha[r]; o3[r] *= alpha[r];
    }
    bf16x16 pa;
#pragma unroll
    for (int j = 0; j < 8; ++j) pa[j] = lp[m16 * 32 + half * 8 + j];
#pragma unroll
    for (int j = 0; j < 8; ++j) pa[8 + j] = lp[m16 * 32 + 16 + half * 8 + j];

    o0 = wmma_bf16(pa, load_vfrag(vc, 0 * 16 + m16, half), o0);
    o1 = wmma_bf16(pa, load_vfrag(vc, 1 * 16 + m16, half), o1);
    o2 = wmma_bf16(pa, load_vfrag(vc, 2 * 16 + m16, half), o2);
    o3 = wmma_bf16(pa, load_vfrag(vc, 3 * 16 + m16, half), o3);

    // refill this buffer with chunk it+2 once every wave is done reading it
    asm volatile("s_wait_dscnt 0" ::: "memory");
    __syncthreads();
    if (wvid == 0 && it + 2 < NCH) {
      tdm_load_tile32x64(lds_off(&kbuf[it & 1][0]), kbase + (size_t)(it + 2) * 32 * HD);
      tdm_load_tile32x64(lds_off(&vbuf[it & 1][0]), vbase + (size_t)(it + 2) * 32 * HD);
    }
  }

  // ---- normalize and store heads-merged fp32 [B,T,H] ----
#pragma unroll
  for (int r = 0; r < 8; ++r) {
    const int q = qtile + r + 8 * half;
    const float inv = 1.0f / fmaxf(lrow[r], 1e-20f);
    float* dst = attn_out + ((size_t)(b * TT + q)) * HH + h * HD;
    dst[0 * 16 + m16] = o0[r] * inv;
    dst[1 * 16 + m16] = o1[r] * inv;
    dst[2 * 16 + m16] = o2[r] * inv;
    dst[3 * 16 + m16] = o3[r] * inv;
  }
}

// ---------------------------------------------------------------------------
// Kernel 5: LayerNorm over H, bf16 out for the final WMMA GEMM.
// ---------------------------------------------------------------------------
__global__ void layernorm_kernel(const float* __restrict__ attn, const float* __restrict__ g,
                                 const float* __restrict__ be, bf16_t* __restrict__ out) {
  const int row = blockIdx.x;
  const int tid = threadIdx.x;
  __shared__ float ssum[256], ssq[256];
  const float* p = attn + (size_t)row * HH;
  float s = 0.f, q = 0.f;
  for (int i = tid; i < HH; i += 256) { const float v = p[i]; s += v; q += v * v; }
  ssum[tid] = s; ssq[tid] = q;
  __syncthreads();
  for (int off = 128; off > 0; off >>= 1) {
    if (tid < off) { ssum[tid] += ssum[tid + off]; ssq[tid] += ssq[tid + off]; }
    __syncthreads();
  }
  const float mu = ssum[0] * (1.0f / HH);
  const float var = ssq[0] * (1.0f / HH) - mu * mu;
  const float inv = rsqrtf(var + 1e-5f);
  for (int i = tid; i < HH; i += 256)
    out[(size_t)row * HH + i] = f2bf((p[i] - mu) * inv * g[i] + be[i]);
}

// ---------------------------------------------------------------------------
// Kernel 6: output projection  y = ln(attn) @ Wout^T  (fp32 out)
// grid (H/256, M/16), block (32,4): 16x64 strip per wave, A reused 4x.
// ---------------------------------------------------------------------------
__global__ void outproj_kernel(const bf16_t* __restrict__ xln, const bf16_t* __restrict__ w,
                               float* __restrict__ out) {
  const int lane = threadIdx.x, m16 = lane & 15, half = lane >> 4;
  const int tileN0 = (blockIdx.x * 4 + threadIdx.y) * 64;
  const int tileM = blockIdx.y * 16;
  floatx8 acc[4] = {};
  for (int k0 = 0; k0 < HH; k0 += 32) {
    const bf16x16 a = frag_from_bf16(xln, HH, tileM + m16, k0, half);
#pragma unroll
    for (int j = 0; j < 4; ++j) {
      bf16x16 b = frag_from_bf16(w, HH, tileN0 + j * 16 + m16, k0, half);
      acc[j] = wmma_bf16(a, b, acc[j]);
    }
  }
#pragma unroll
  for (int j = 0; j < 4; ++j)
#pragma unroll
    for (int r = 0; r < 8; ++r)
      out[(size_t)(tileM + r + 8 * half) * HH + tileN0 + j * 16 + m16] = acc[j][r];
}

extern "C" void kernel_launch(void* const* d_in, const int* in_sizes, int n_in,
                              void* d_out, int out_size, void* d_ws, size_t ws_size,
                              hipStream_t stream) {
  (void)in_sizes; (void)n_in; (void)out_size; (void)ws_size;
  const float*   x       = (const float*)d_in[0];
  const float*   bias    = (const float*)d_in[1];
  const float*   lw      = (const float*)d_in[2];
  const uint8_t* kpm     = (const uint8_t*)d_in[3];
  const int*     outcell = (const int*)d_in[4];
  const uint8_t* em      = (const uint8_t*)d_in[5];
  const float*   wq      = (const float*)d_in[6];
  const float*   wk      = (const float*)d_in[7];
  const float*   wv      = (const float*)d_in[8];
  const float*   wout    = (const float*)d_in[9];
  const float*   ln_g    = (const float*)d_in[10];
  const float*   ln_b    = (const float*)d_in[11];
  float* out = (float*)d_out;

  char* ws = (char*)d_ws;
  size_t off = 0;
  float*  qf     = (float*)(ws + off);  off += (size_t)BB * TT * HH * 4;
  float*  kf     = (float*)(ws + off);  off += (size_t)BB * TT * HH * 4;
  float*  vf     = (float*)(ws + off);  off += (size_t)BB * TT * HH * 4;
  bf16_t* qh     = (bf16_t*)(ws + off); off += (size_t)BB * NHEAD * TT * HD * 2;
  bf16_t* kh     = (bf16_t*)(ws + off); off += (size_t)BB * NHEAD * SS * HD * 2;
  bf16_t* vh     = (bf16_t*)(ws + off); off += (size_t)BB * NHEAD * SS * HD * 2;
  float*  attnb  = (float*)(ws + off);  off += (size_t)BB * TT * HH * 4;
  bf16_t* xln    = (bf16_t*)(ws + off); off += (size_t)BB * TT * HH * 2;
  bf16_t* xbf    = (bf16_t*)(ws + off); off += (size_t)BB * TT * HH * 2;
  bf16_t* wqbf   = (bf16_t*)(ws + off); off += (size_t)HH * HH * 2;
  bf16_t* wkbf   = (bf16_t*)(ws + off); off += (size_t)HH * HH * 2;
  bf16_t* wvbf   = (bf16_t*)(ws + off); off += (size_t)HH * HH * 2;
  bf16_t* woutbf = (bf16_t*)(ws + off); off += (size_t)HH * HH * 2;

  // one-time bf16 conversions (keeps per-tile convert VALU out of the GEMMs)
  f32_to_bf16_kernel<<<(BB * TT * HH) / 1024, 256, 0, stream>>>(x, xbf, BB * TT * HH);
  f32_to_bf16_kernel<<<(HH * HH) / 1024, 256, 0, stream>>>(wq, wqbf, HH * HH);
  f32_to_bf16_kernel<<<(HH * HH) / 1024, 256, 0, stream>>>(wk, wkbf, HH * HH);
  f32_to_bf16_kernel<<<(HH * HH) / 1024, 256, 0, stream>>>(wv, wvbf, HH * HH);
  f32_to_bf16_kernel<<<(HH * HH) / 1024, 256, 0, stream>>>(wout, woutbf, HH * HH);

  qkv_gemm_kernel<<<dim3(HH / 256, (BB * TT) / 16, 3), dim3(32, 4), 0, stream>>>(
      xbf, wqbf, wkbf, wvbf, qf, kf, vf);
  rope_q_kernel<<<(BB * TT * HH) / 256, 256, 0, stream>>>(qf, qh);
  rope_kv_kernel<<<(BB * SS * HH) / 256, 256, 0, stream>>>(kf, vf, outcell, kh, vh);
  attention_kernel<<<dim3(TT / 64, NHEAD, BB), dim3(32, 4), 0, stream>>>(
      qh, kh, vh, bias, lw, kpm, em, attnb);
  layernorm_kernel<<<BB * TT, 256, 0, stream>>>(attnb, ln_g, ln_b, xln);
  outproj_kernel<<<dim3(HH / 256, (BB * TT) / 16), dim3(32, 4), 0, stream>>>(
      xln, woutbf, out);
}
